// _EquivBlock_40407052321387
// MI455X (gfx1250) — matrix operand order repaired and checked by reading, not compiled, not run on hardware
//
#include <hip/hip_runtime.h>
#include <math.h>

// ---------------------------------------------------------------------------
// Problem constants (from reference)
// ---------------------------------------------------------------------------
#define NB    2
#define NN    8000
#define NE    32000
#define CIN   16
#define MUL0  32
#define MUL1  16
#define MUL2  8
#define DIMO  120          // 32 + 3*16 + 5*8
#define HID   32
#define WNUM  896
#define NT    (NB*NN)      // 16000 nodes total
#define ET    (NB*NE)      // 64000 edges total

#define EPW   128          // edges per workgroup
#define TH_EDGE 128        // 4 waves of 32

// T matrix: 512 (=HID*CIN) x 64 (56 cols used, padded) in f16, pre-swizzled
// into B-fragment order: halfs indexed by ((c*4+n)*32 + lane)*16 + j*2 + h
#define T_HALFS 32768      // 64 KB

typedef __attribute__((ext_vector_type(16))) _Float16 v16h;
typedef __attribute__((ext_vector_type(8)))  _Float16 v8h;
typedef __attribute__((ext_vector_type(2)))  _Float16 v2h;
typedef __attribute__((ext_vector_type(8)))  float    v8f;

// TDM descriptor word groups
typedef __attribute__((ext_vector_type(4))) unsigned int tdm_v4u;
typedef __attribute__((ext_vector_type(8))) int          tdm_v8i;
typedef __attribute__((ext_vector_type(4))) int          tdm_v4i;

// scale = (1/sqrt(CIN)) * (1/sqrt(HID)) = 0.25 / sqrt(32)
#define T_SCALE 0.04419417382415922f

// ---------------------------------------------------------------------------
// Kernel 0: repack W2 (f32, HIDxWNUM) -> swizzled f16 B-fragment image
// B-fragment layout (16-bit B 32x16, wave32): lane%16 = column N,
// lanes 0-15 hold K=0..7,16..23 (VGPR j<4 -> K=2j,2j+1; j>=4 -> K=2j+8,2j+9), lanes 16-31 the same +8.
// kk = c*32 + Klocal; k = kk>>4 (HID index), u = kk&15 (CIN index).
// col<32  -> W2[k, u*32+col]          (0e path, MUL0)
// col<48  -> W2[k, 512 + u*16+col-32] (1o path, MUL1)
// col<56  -> W2[k, 768 + u*8 +col-48] (2e path, MUL2)
// else 0 (padding)
// ---------------------------------------------------------------------------
__global__ void k_repack_w2(const float* __restrict__ W2,
                            _Float16* __restrict__ Timg) {
    int flat = blockIdx.x * blockDim.x + threadIdx.x;
    if (flat >= T_HALFS) return;
    int h  = flat & 1;
    int j  = (flat >> 1) & 7;
    int L  = (flat >> 4) & 31;
    int n  = (flat >> 9) & 3;
    int c  = flat >> 11;              // 0..15
    int Ncol = L & 15;
    int hi   = L >> 4;
    int Klocal = (j < 4 ? 0 : 16) + (hi ? 8 : 0) + 2 * (j & 3) + h;
    int kk = c * 32 + Klocal;
    int k  = kk >> 4;
    int u  = kk & 15;
    int col = n * 16 + Ncol;
    float v = 0.0f;
    if (col < 32)       v = W2[k * WNUM + u * 32 + col];
    else if (col < 48)  v = W2[k * WNUM + 512 + u * 16 + (col - 32)];
    else if (col < 56)  v = W2[k * WNUM + 768 + u * 8  + (col - 48)];
    Timg[flat] = (_Float16)(v * T_SCALE);
}

// ---------------------------------------------------------------------------
// Kernel 1: per-node prep.  h_in = h @ W_lin * 0.25 + b_lin  (16->16)
// acc[:, 0:32] = h @ W_res * 0.25 + b_res ; acc[:, 32:120] = 0
// ---------------------------------------------------------------------------
__global__ void k_node_prep(const float* __restrict__ h,
                            const float* __restrict__ Wlin,
                            const float* __restrict__ blin,
                            const float* __restrict__ Wres,
                            const float* __restrict__ bres,
                            float* __restrict__ hin,
                            float* __restrict__ accb) {
    int nid = blockIdx.x * blockDim.x + threadIdx.x;
    if (nid >= NT) return;
    float hv[CIN];
#pragma unroll
    for (int c = 0; c < CIN; ++c) hv[c] = h[nid * CIN + c];
#pragma unroll
    for (int c2 = 0; c2 < CIN; ++c2) {
        float s = 0.f;
#pragma unroll
        for (int c = 0; c < CIN; ++c) s += hv[c] * Wlin[c * CIN + c2];
        hin[nid * CIN + c2] = s * 0.25f + blin[c2];
    }
    float* ap = accb + (size_t)nid * DIMO;
#pragma unroll
    for (int w = 0; w < MUL0; ++w) {
        float s = 0.f;
#pragma unroll
        for (int c = 0; c < CIN; ++c) s += hv[c] * Wres[c * MUL0 + w];
        ap[w] = s * 0.25f + bres[w];
    }
#pragma unroll
    for (int i = MUL0; i < DIMO; ++i) ap[i] = 0.f;
}

// ---------------------------------------------------------------------------
// Kernel 2: edge kernel.  Per WG: 128 edges, 4 waves x 32 edges.
// G(e, kk) = hr[e, kk>>4] * hsrc[e, kk&15]  (formed in f16)
// C(128 x 64) += G(128 x 512) @ T(512 x 64) via v_wmma_f32_16x16x32_f16
// then scatter msg = C * Y into acc buffer with atomics.
// T staged global->LDS with the Tensor Data Mover (one 64KB DMA per WG).
// ---------------------------------------------------------------------------
__global__ void __launch_bounds__(TH_EDGE)
k_edges(const float* __restrict__ hin,
        const _Float16* __restrict__ Timg,
        const int* __restrict__ e_src,
        const int* __restrict__ e_dst,
        const float* __restrict__ e_attr,
        const float* __restrict__ W1,
        float* __restrict__ accb) {
    __shared__ __align__(32) _Float16 Tl[T_HALFS];     // 64 KB
    __shared__ __align__(16) _Float16 hs_l[EPW][CIN];  // 4 KB
    __shared__ __align__(4)  _Float16 hr_l[EPW][HID];  // 8 KB
    __shared__ float Y_l[EPW][12];                     // 6 KB
    __shared__ int   dst_l[EPW];                       // 0.5 KB

    const int tid = threadIdx.x;

    // ---- stage T into LDS: TDM async tensor DMA (64 KB contiguous) ----
#if __has_builtin(__builtin_amdgcn_tensor_load_to_lds) && \
    __has_builtin(__builtin_amdgcn_s_wait_tensorcnt)
    {
        // Uniform scalar branch: only wave 0 issues the DMA (TDM ignores
        // EXEC, so an exec-mask guard would issue one DMA per wave).
        int wv = __builtin_amdgcn_readfirstlane(tid >> 5);
        if (wv == 0) {
            unsigned long long ga = (unsigned long long)(uintptr_t)Timg;
            unsigned int lds_off = (unsigned int)(uintptr_t)(void*)Tl;
            // D# group0: count=1 | lds_addr | global_addr | type=2
            tdm_v4u g0;
            g0[0] = 1u;                                   // count=1, user mode
            g0[1] = lds_off;                              // lds_addr [63:32]
            g0[2] = (unsigned int)ga;                     // global_addr[31:0]
            g0[3] = (unsigned int)(ga >> 32) | (2u << 30);// ga[56:32] | type=2
            // D# group1: 1 row of 16384 dwords (data_size=4B)
            tdm_v8i g1;
            g1[0] = (int)(2u << 16);          // data_size=2 (4B), mask=0
            g1[1] = (int)(16384u << 16);      // tensor_dim0[15:0]=16384
            g1[2] = (int)(1u << 16);          // tensor_dim0[31:16]=0, dim1=1
            g1[3] = (int)(16384u << 16);      // tile_dim0=16384
            g1[4] = 1;                        // tile_dim1=1, tile_dim2=0
            g1[5] = 16384;                    // tensor_dim0_stride
            g1[6] = 0;
            g1[7] = 0;
            tdm_v4i g2 = (tdm_v4i)0;
            tdm_v4i g3 = (tdm_v4i)0;
#if defined(__clang_major__) && (__clang_major__ >= 23)
            tdm_v8i g4 = (tdm_v8i)0;
            __builtin_amdgcn_tensor_load_to_lds(g0, g1, g2, g3, g4, 0);
#else
            __builtin_amdgcn_tensor_load_to_lds(g0, g1, g2, g3, 0);
#endif
        }
        __builtin_amdgcn_s_wait_tensorcnt(0);
    }
#else
    {
        const uint4* ts = (const uint4*)Timg;
        uint4* td = (uint4*)Tl;
        for (int i = tid; i < T_HALFS / 8; i += TH_EDGE) td[i] = ts[i];
    }
#endif

    // ---- per-edge prep: one edge per thread ----
    {
        long eg = (long)blockIdx.x * EPW + tid;
        bool valid = (eg < (long)ET);
        long ec = valid ? eg : 0;
        int b = (int)(ec / NE);
        int src = e_src[ec];
        int dst = e_dst[ec];
        int nsrc = b * NN + src;
        const float* hp = hin + (size_t)nsrc * CIN;
#pragma unroll
        for (int c = 0; c < CIN; ++c) hs_l[tid][c] = (_Float16)hp[c];

        float x = e_attr[ec * 3 + 0];
        float y = e_attr[ec * 3 + 1];
        float z = e_attr[ec * 3 + 2];
        float rn = sqrtf(x * x + y * y + z * z) + 1e-8f;
        float ir = 1.0f / rn;
        float xh = x * ir, yh = y * ir, zh = z * ir;
        const float s3 = 1.7320508075688772f;
        const float s5 = 2.23606797749979f;
        const float s15 = 3.872983346207417f;
        Y_l[tid][0] = 1.0f;
        Y_l[tid][1] = s3 * yh;
        Y_l[tid][2] = s3 * zh;
        Y_l[tid][3] = s3 * xh;
        Y_l[tid][4] = s15 * xh * yh;
        Y_l[tid][5] = s15 * yh * zh;
        Y_l[tid][6] = 0.5f * s5 * (3.0f * zh * zh - 1.0f);
        Y_l[tid][7] = s15 * xh * zh;
        Y_l[tid][8] = 0.5f * s15 * (xh * xh - yh * yh);

#pragma unroll
        for (int k = 0; k < HID; ++k) {
            float t = rn * W1[k];
            hr_l[tid][k] = (_Float16)(t / (1.0f + __expf(-t)));   // silu
        }
        dst_l[tid] = valid ? (b * NN + dst) : -1;
    }
    __syncthreads();

    // ---- WMMA phase: wave handles 32 edges = 2 M-tiles, 4 N-tiles ----
    const int wave = tid >> 5;
    const int lane = tid & 31;
    const int lo16 = lane & 15;
    const int hi   = lane >> 4;            // 0: K 0-7/16-23, u 0-7 ; 1: u 8-15
    const int row0 = wave * 32 + lo16;     // M-tile 0 edge (local)
    const int row1 = row0 + 16;            // M-tile 1 edge (local)
    const int u0   = hi * 8;

    // per-lane h_src slice (u = u0..u0+7), f16, one 16B LDS read each
    v8h hsA = *(const v8h*)&hs_l[row0][u0];
    v8h hsB = *(const v8h*)&hs_l[row1][u0];

    v8f acc[2][4];
#pragma unroll
    for (int m = 0; m < 2; ++m)
#pragma unroll
        for (int n = 0; n < 4; ++n) acc[m][n] = (v8f)0.0f;

    const v16h* Tv = (const v16h*)Tl;      // 32B per lane-fragment slice

#pragma unroll 4
    for (int c = 0; c < 16; ++c) {
        // hr pair (k = 2c, 2c+1) needed by every lane for this K-chunk
        v2h ha = *(const v2h*)&hr_l[row0][2 * c];
        v2h hb = *(const v2h*)&hr_l[row1][2 * c];
        // Build A fragments (16-bit A 16x32 layout), all-f16 math:
        //  VGPR j<4 : K = (hi?8:0)+2j+h   -> k=2c,   u = u0+2j+h
        //  VGPR j>=4: K = 16+(hi?8:0)+... -> k=2c+1, u = u0+2(j-4)+h
        v16h A0, A1;
#pragma unroll
        for (int j = 0; j < 4; ++j) {
            A0[2 * j]         = ha[0] * hsA[2 * j];
            A0[2 * j + 1]     = ha[0] * hsA[2 * j + 1];
            A0[8 + 2 * j]     = ha[1] * hsA[2 * j];
            A0[8 + 2 * j + 1] = ha[1] * hsA[2 * j + 1];
            A1[2 * j]         = hb[0] * hsB[2 * j];
            A1[2 * j + 1]     = hb[0] * hsB[2 * j + 1];
            A1[8 + 2 * j]     = hb[1] * hsB[2 * j];
            A1[8 + 2 * j + 1] = hb[1] * hsB[2 * j + 1];
        }
#pragma unroll
        for (int n = 0; n < 4; ++n) {
            v16h Bf = Tv[(c * 4 + n) * 32 + lane];
            acc[0][n] = __builtin_amdgcn_wmma_f32_16x16x32_f16(
                false, A0, false, Bf, (short)0, acc[0][n], false, false);
            acc[1][n] = __builtin_amdgcn_wmma_f32_16x16x32_f16(
                false, A1, false, Bf, (short)0, acc[1][n], false, false);
        }
    }

    // ---- epilogue: scatter through Y with atomics ----
    // D layout: lane L, reg r holds element (M = r + 8*hi, N = L&15)
#pragma unroll
    for (int mt = 0; mt < 2; ++mt) {
        int ebase = wave * 32 + mt * 16;
#pragma unroll
        for (int r = 0; r < 8; ++r) {
            int M = r + hi * 8;
            int el = ebase + M;
            int d = dst_l[el];
            if (d < 0) continue;
            float* ap = accb + (size_t)d * DIMO;
            const float* Ye = Y_l[el];
#pragma unroll
            for (int n = 0; n < 4; ++n) {
                int col = n * 16 + lo16;
                float v = acc[mt][n][r];
                if (col < 32) {
                    atomicAdd(ap + col, v);                      // Y0 == 1
                } else if (col < 48) {
                    int w = col - 32;
                    int base = MUL0 + w * 3;
                    atomicAdd(ap + base + 0, v * Ye[1]);
                    atomicAdd(ap + base + 1, v * Ye[2]);
                    atomicAdd(ap + base + 2, v * Ye[3]);
                } else if (col < 56) {
                    int w = col - 48;
                    int base = MUL0 + 3 * MUL1 + w * 5;
#pragma unroll
                    for (int m = 0; m < 5; ++m)
                        atomicAdd(ap + base + m, v * Ye[4 + m]);
                }
            }
        }
    }
}

// ---------------------------------------------------------------------------
// Kernel 3: NormActivation per irrep block: out = x * sigmoid(||x||_irrep)
// ---------------------------------------------------------------------------
__global__ void k_norm_act(const float* __restrict__ accb,
                           float* __restrict__ out) {
    int nid = blockIdx.x * blockDim.x + threadIdx.x;
    if (nid >= NT) return;
    const float* xp = accb + (size_t)nid * DIMO;
    float* op = out + (size_t)nid * DIMO;
#pragma unroll
    for (int i = 0; i < MUL0; ++i) {
        float x = xp[i];
        float nn = sqrtf(x * x + 1e-12f);
        op[i] = x / (1.0f + __expf(-nn));
    }
#pragma unroll
    for (int w = 0; w < MUL1; ++w) {
        int base = MUL0 + w * 3;
        float a = xp[base], b = xp[base + 1], cc = xp[base + 2];
        float nn = sqrtf(a * a + b * b + cc * cc + 1e-12f);
        float f = 1.0f / (1.0f + __expf(-nn));
        op[base] = a * f; op[base + 1] = b * f; op[base + 2] = cc * f;
    }
#pragma unroll
    for (int w = 0; w < MUL2; ++w) {
        int base = MUL0 + 3 * MUL1 + w * 5;
        float s = 1e-12f;
#pragma unroll
        for (int m = 0; m < 5; ++m) { float x = xp[base + m]; s += x * x; }
        float nn = sqrtf(s);
        float f = 1.0f / (1.0f + __expf(-nn));
#pragma unroll
        for (int m = 0; m < 5; ++m) op[base + m] = xp[base + m] * f;
    }
}

// ---------------------------------------------------------------------------
// Launch
// ---------------------------------------------------------------------------
extern "C" void kernel_launch(void* const* d_in, const int* in_sizes, int n_in,
                              void* d_out, int out_size, void* d_ws, size_t ws_size,
                              hipStream_t stream) {
    (void)in_sizes; (void)n_in; (void)out_size; (void)ws_size;
    const float* h      = (const float*)d_in[0];
    const int*   e_src  = (const int*)  d_in[1];
    const int*   e_dst  = (const int*)  d_in[2];
    const float* e_attr = (const float*)d_in[3];
    const float* W_lin  = (const float*)d_in[4];
    const float* b_lin  = (const float*)d_in[5];
    const float* W1     = (const float*)d_in[6];
    const float* W2     = (const float*)d_in[7];
    const float* W_res  = (const float*)d_in[8];
    const float* b_res  = (const float*)d_in[9];
    float* out = (float*)d_out;

    char* ws = (char*)d_ws;
    float*    hin  = (float*)ws;                               // NT*16*4  = 1.0 MB
    float*    accb = (float*)(ws + (size_t)NT * CIN * 4);      // NT*120*4 = 7.3 MB
    _Float16* Timg = (_Float16*)(ws + (size_t)NT * CIN * 4
                                    + (size_t)NT * DIMO * 4);  // 64 KB

    k_repack_w2<<<T_HALFS / 256, 256, 0, stream>>>(W2, Timg);
    k_node_prep<<<(NT + 255) / 256, 256, 0, stream>>>(h, W_lin, b_lin,
                                                      W_res, b_res, hin, accb);
    k_edges<<<ET / EPW, TH_EDGE, 0, stream>>>(hin, Timg, e_src, e_dst,
                                              e_attr, W1, accb);
    k_norm_act<<<(NT + 255) / 256, 256, 0, stream>>>(accb, out);
}
